// ResidualAttentionBlock_14800457302228
// MI455X (gfx1250) — matrix operand order repaired and checked by reading, not compiled, not run on hardware
//
#include <hip/hip_runtime.h>
#include <hip/hip_bf16.h>

// ---------------------------------------------------------------------------
// MI455X (gfx1250) implementation: bf16 WMMA everywhere.
//   S = 1500 tokens (padded to 1504 = 94 tiles of 16), C = 1280, H=20, D=64.
// GEMMs: 32x64 output tile per wave, register double-buffered fragment loads;
// K/N/MODE are template constants so all tile offsets fold into the b128
// immediate-offset field. Attention: flash-style with the softmax row-sum
// computed as P @ ones via an extra WMMA (no ds_bpermute reduction for l).
// ---------------------------------------------------------------------------

typedef __attribute__((ext_vector_type(16))) __bf16 v16bf;
typedef __attribute__((ext_vector_type(8)))  float  v8f;
typedef __attribute__((ext_vector_type(4)))  unsigned int u32x4;

union ABFrag { v16bf v; u32x4 q[2]; };

#define SEQ     1500
#define SEQ_PAD 1504
#define NSTATE  1280
#define NMLP    5120
#define NHEAD   20
#define DHEAD   64
#define MTILES  94      // SEQ_PAD / 16
#define KTILES  47      // SEQ_PAD / 32

enum { MODE_BF16 = 0, MODE_BF16_T = 1, MODE_GELU_BF16 = 2, MODE_F32_RES = 3 };

// ---------------------------------------------------------------------------
// fp32 [K,N] -> bf16 [N,K] transpose-convert (weights; B-fragments become two
// contiguous 16B loads per lane).
// ---------------------------------------------------------------------------
__global__ __launch_bounds__(256) void convert_w_t(const float* __restrict__ W,
                                                   __bf16* __restrict__ Wt,
                                                   int K, int N) {
  size_t tid = (size_t)blockIdx.x * 256 + threadIdx.x;
  size_t total = (size_t)K * N;
  if (tid >= total) return;
  size_t kk = tid % (size_t)K;
  size_t nn = tid / (size_t)K;
  Wt[tid] = (__bf16)W[kk * (size_t)N + nn];
}

// ---------------------------------------------------------------------------
// LayerNorm (fp32 in -> bf16 out), rows >= SEQ zero-filled (padding rows).
// ---------------------------------------------------------------------------
__global__ __launch_bounds__(256) void ln_kernel(const float* __restrict__ x,
                                                 const float* __restrict__ gm,
                                                 const float* __restrict__ bt,
                                                 __bf16* __restrict__ out) {
  const int C = NSTATE;
  int row = blockIdx.x;
  __bf16* orow = out + (size_t)row * C;
  if (row >= SEQ) {
    for (int c = threadIdx.x; c < C; c += 256) orow[c] = (__bf16)0.0f;
    return;
  }
  const float* xr = x + (size_t)row * C;
  float s = 0.f, s2 = 0.f;
  for (int c = threadIdx.x; c < C; c += 256) {
    float v = xr[c];
    s += v; s2 += v * v;
  }
  __shared__ float rs[256], rs2[256];
  rs[threadIdx.x] = s; rs2[threadIdx.x] = s2;
  __syncthreads();
  for (int o = 128; o > 0; o >>= 1) {
    if ((int)threadIdx.x < o) {
      rs[threadIdx.x]  += rs[threadIdx.x + o];
      rs2[threadIdx.x] += rs2[threadIdx.x + o];
    }
    __syncthreads();
  }
  float mu   = rs[0] * (1.0f / C);
  float var  = rs2[0] * (1.0f / C) - mu * mu;
  float rstd = rsqrtf(var + 1e-5f);
  for (int c = threadIdx.x; c < C; c += 256)
    orow[c] = (__bf16)((xr[c] - mu) * rstd * gm[c] + bt[c]);
}

// ---------------------------------------------------------------------------
// WMMA bf16 GEMM:  out = act(A[S_PAD,K] @ Bt[N,K]^T + bias)
// One wave computes a 32x64 output tile (2 M-tiles x 4 N-tiles = 8 WMMAs per
// 32-wide K step). Fragments double-buffered: loads for step s+1 issue before
// the WMMAs of step s. K/N/MODE compile-time so tile offsets are immediates.
// ---------------------------------------------------------------------------
template <int K, int N, int MODE>
__global__ __launch_bounds__(128, 1)
void gemm_bf16_wmma(const __bf16* __restrict__ A, const __bf16* __restrict__ Bt,
                    const float* __restrict__ bias, void* __restrict__ out,
                    const float* __restrict__ res) {
  int lane = threadIdx.x & 31;
  int g    = lane >> 4;       // half-wave group
  int ln   = lane & 15;
  int wave = threadIdx.x >> 5;
  int m0   = blockIdx.x * 32;                      // 32 rows per wave
  int nb   = blockIdx.y * 256 + wave * 64;         // 64 output cols per wave

  // single base pointer per matrix; m/n tile offsets are template constants
  const __bf16* abase = A  + (size_t)(m0 + ln) * K + 8 * g;
  const __bf16* bbase = Bt + (size_t)(nb + ln) * K + 8 * g;

  v8f zero = {0.f, 0.f, 0.f, 0.f, 0.f, 0.f, 0.f, 0.f};
  v8f acc[2][4] = {{zero, zero, zero, zero}, {zero, zero, zero, zero}};

  ABFrag a[2][2];     // [buffer][m-tile]
  ABFrag b[2][4];     // [buffer][n-tile]

  auto load_step = [&](int buf, int k0) {
#pragma unroll
    for (int mi = 0; mi < 2; ++mi) {
      a[buf][mi].q[0] = *(const u32x4*)(abase + mi * 16 * K + k0);
      a[buf][mi].q[1] = *(const u32x4*)(abase + mi * 16 * K + k0 + 16);
    }
#pragma unroll
    for (int t = 0; t < 4; ++t) {
      b[buf][t].q[0] = *(const u32x4*)(bbase + t * 16 * K + k0);
      b[buf][t].q[1] = *(const u32x4*)(bbase + t * 16 * K + k0 + 16);
    }
  };
  auto wmma_step = [&](int buf) {
#pragma unroll
    for (int mi = 0; mi < 2; ++mi)
#pragma unroll
      for (int t = 0; t < 4; ++t)
        acc[mi][t] = __builtin_amdgcn_wmma_f32_16x16x32_bf16(
            false, a[buf][mi].v, false, b[buf][t].v, (short)0, acc[mi][t],
            false, false);
  };

  constexpr int steps = K / 32;
  load_step(0, 0);
  int k0 = 32;
  for (int s = 0; s < steps - 1; ++s, k0 += 32) {
    int cur = s & 1;
    load_step(cur ^ 1, k0);   // prefetch next step before consuming current
    wmma_step(cur);
  }
  wmma_step((steps - 1) & 1);

#pragma unroll
  for (int mi = 0; mi < 2; ++mi)
#pragma unroll
    for (int t = 0; t < 4; ++t) {
      int col = nb + t * 16 + ln;
      float bval = bias ? bias[col] : 0.0f;
#pragma unroll
      for (int r = 0; r < 8; ++r) {
        int row = m0 + mi * 16 + 8 * g + r;        // D layout: row = 8g+r
        float v = acc[mi][t][r] + bval;
        if (MODE == MODE_BF16) {
          ((__bf16*)out)[(size_t)row * N + col] = (__bf16)v;
        } else if (MODE == MODE_BF16_T) {
          ((__bf16*)out)[(size_t)col * SEQ_PAD + row] = (__bf16)v;
        } else if (MODE == MODE_GELU_BF16) {
          float gl = 0.5f * v * (1.0f + erff(v * 0.70710678118f));
          ((__bf16*)out)[(size_t)row * N + col] = (__bf16)gl;
        } else { // MODE_F32_RES
          if (row < SEQ)
            ((float*)out)[(size_t)row * N + col] =
                res[(size_t)row * N + col] + v;
        }
      }
    }
}

// ---------------------------------------------------------------------------
// Flash attention: one wave per (head, 16-query tile); streams 32-key tiles,
// online softmax. Row max uses a shfl_xor butterfly; the row SUM is computed
// on the matrix cores as P @ ones (one extra WMMA per tile) so the l-state
// lives in a D-layout accumulator and rescales with the same alpha[r] slots.
// q,k: bf16 [S_PAD, C];  vt: bf16 [C, S_PAD] (transposed);  ctx: bf16 [S_PAD,C]
// ---------------------------------------------------------------------------
__global__ __launch_bounds__(128, 1)
void attn_kernel(const __bf16* __restrict__ q, const __bf16* __restrict__ k,
                 const __bf16* __restrict__ vt, __bf16* __restrict__ ctx) {
  const int C = NSTATE;
  int wave = threadIdx.x >> 5;
  int lane = threadIdx.x & 31;
  int g = lane >> 4, ln = lane & 15;
  int wid  = blockIdx.x * 4 + wave;   // 0..1879
  int head = wid / MTILES;
  int qt   = wid % MTILES;

  __shared__ __bf16 pshared[4][16 * 32];
  __bf16* P = pshared[wave];

  // Q fragments for d = 0..31 and d = 32..63
  ABFrag qa[2];
  {
    const __bf16* qr = q + (size_t)(qt * 16 + ln) * C + head * DHEAD;
    qa[0].q[0] = *(const u32x4*)(qr + 8 * g);
    qa[0].q[1] = *(const u32x4*)(qr + 16 + 8 * g);
    qa[1].q[0] = *(const u32x4*)(qr + 32 + 8 * g);
    qa[1].q[1] = *(const u32x4*)(qr + 48 + 8 * g);
  }

  // constant all-ones B fragment: l = P @ ones lands row sums in D layout
  ABFrag ones;
#pragma unroll
  for (int i = 0; i < 16; ++i) ones.v[i] = (__bf16)1.0f;

  v8f zero = {0.f, 0.f, 0.f, 0.f, 0.f, 0.f, 0.f, 0.f};
  v8f o[4] = {zero, zero, zero, zero};
  v8f lacc = zero;                       // running softmax denominator per row
  float mrow[8];
#pragma unroll
  for (int r = 0; r < 8; ++r) mrow[r] = -1e30f;

  for (int kt = 0; kt < KTILES; ++kt) {
    int key0 = kt * 32;
    // ---- scores: 16 queries x 32 keys, two 16x16 D tiles -----------------
    v8f s[2] = {zero, zero};
#pragma unroll
    for (int half = 0; half < 2; ++half) {
      const __bf16* kr = k + (size_t)(key0 + half * 16 + ln) * C + head * DHEAD;
#pragma unroll
      for (int st = 0; st < 2; ++st) {
        ABFrag kb;                       // B = K^T fragment, table order in d
        kb.q[0] = *(const u32x4*)(kr + 32 * st + 8 * g);
        kb.q[1] = *(const u32x4*)(kr + 32 * st + 8 * g + 16);
        s[half] = __builtin_amdgcn_wmma_f32_16x16x32_bf16(
            false, qa[st].v, false, kb.v, (short)0, s[half], false, false);
      }
    }
    // scale by 1/sqrt(64), mask padded keys (whole lane = one key column)
#pragma unroll
    for (int half = 0; half < 2; ++half) {
      bool valid = (key0 + half * 16 + ln) < SEQ;
#pragma unroll
      for (int r = 0; r < 8; ++r) {
        float v = s[half][r] * 0.125f;
        s[half][r] = valid ? v : -1e30f;
      }
    }
    // ---- row max (row = 8g+r lives across the 16 lanes of group g) ------
    float tmax[8];
#pragma unroll
    for (int r = 0; r < 8; ++r) tmax[r] = fmaxf(s[0][r], s[1][r]);
#pragma unroll
    for (int off = 8; off > 0; off >>= 1)
#pragma unroll
      for (int r = 0; r < 8; ++r)
        tmax[r] = fmaxf(tmax[r], __shfl_xor(tmax[r], off, 16));
    float alpha[8];
#pragma unroll
    for (int r = 0; r < 8; ++r) {
      float mn = fmaxf(mrow[r], tmax[r]);
      alpha[r] = __expf(mrow[r] - mn);
      mrow[r] = mn;
    }
#pragma unroll
    for (int half = 0; half < 2; ++half)
#pragma unroll
      for (int r = 0; r < 8; ++r)
        s[half][r] = __expf(s[half][r] - mrow[r]);
    // rescale running state (o and l share the per-row alpha slots)
#pragma unroll
    for (int r = 0; r < 8; ++r) lacc[r] *= alpha[r];
#pragma unroll
    for (int t = 0; t < 4; ++t)
#pragma unroll
      for (int r = 0; r < 8; ++r) o[t][r] *= alpha[r];

    // ---- P (D layout) -> LDS -> A-matrix fragment ------------------------
#pragma unroll
    for (int half = 0; half < 2; ++half)
#pragma unroll
      for (int r = 0; r < 8; ++r)
        P[(8 * g + r) * 32 + half * 16 + ln] = (__bf16)s[half][r];
    ABFrag pa;   // per-wave LDS tile; DS ops are in-order within a wave
    pa.q[0] = *(const u32x4*)(P + ln * 32 + 8 * g);
    pa.q[1] = *(const u32x4*)(P + ln * 32 + 16 + 8 * g);

    // ---- l += P @ ones : row sums via matrix core (no shuffle reduce) ----
    lacc = __builtin_amdgcn_wmma_f32_16x16x32_bf16(
        false, pa.v, false, ones.v, (short)0, lacc, false, false);

    // ---- o += P @ V  (4 d-tiles of 16) -----------------------------------
#pragma unroll
    for (int t = 0; t < 4; ++t) {
      const __bf16* vr =
          vt + (size_t)(head * DHEAD + t * 16 + ln) * SEQ_PAD + key0 + 8 * g;
      ABFrag vb;
      vb.q[0] = *(const u32x4*)(vr);
      vb.q[1] = *(const u32x4*)(vr + 16);
      o[t] = __builtin_amdgcn_wmma_f32_16x16x32_bf16(
          false, pa.v, false, vb.v, (short)0, o[t], false, false);
    }
  }

  // ---- normalize and store context --------------------------------------
  float inv[8];
#pragma unroll
  for (int r = 0; r < 8; ++r) inv[r] = 1.0f / lacc[r];
#pragma unroll
  for (int t = 0; t < 4; ++t)
#pragma unroll
    for (int r = 0; r < 8; ++r) {
      int row = qt * 16 + 8 * g + r;
      int col = head * DHEAD + t * 16 + ln;
      ctx[(size_t)row * C + col] = (__bf16)(o[t][r] * inv[r]);
    }
}

// ---------------------------------------------------------------------------
extern "C" void kernel_launch(void* const* d_in, const int* in_sizes, int n_in,
                              void* d_out, int out_size, void* d_ws,
                              size_t ws_size, hipStream_t stream) {
  const float* x    = (const float*)d_in[0];
  const float* Wq   = (const float*)d_in[1];
  const float* bq   = (const float*)d_in[2];
  const float* Wk   = (const float*)d_in[3];
  const float* Wv   = (const float*)d_in[4];
  const float* bv   = (const float*)d_in[5];
  const float* Wo   = (const float*)d_in[6];
  const float* bo   = (const float*)d_in[7];
  const float* ln1g = (const float*)d_in[8];
  const float* ln1b = (const float*)d_in[9];
  const float* ln2g = (const float*)d_in[10];
  const float* ln2b = (const float*)d_in[11];
  const float* W1   = (const float*)d_in[12];
  const float* b1   = (const float*)d_in[13];
  const float* W2   = (const float*)d_in[14];
  const float* b2   = (const float*)d_in[15];
  float* out = (float*)d_out;

  // workspace carve-up (256B aligned)
  char* ws = (char*)d_ws;
  size_t off = 0;
  auto alloc = [&](size_t bytes) -> void* {
    void* p = ws + off;
    off = (off + bytes + 255) & ~(size_t)255;
    return p;
  };
  const size_t WCC = (size_t)NSTATE * NSTATE;   // 1280*1280
  const size_t WCM = (size_t)NSTATE * NMLP;     // 1280*5120
  const size_t ACT = (size_t)SEQ_PAD * NSTATE;  // padded activations
  __bf16* Wq_t = (__bf16*)alloc(WCC * 2);
  __bf16* Wk_t = (__bf16*)alloc(WCC * 2);
  __bf16* Wv_t = (__bf16*)alloc(WCC * 2);
  __bf16* Wo_t = (__bf16*)alloc(WCC * 2);
  __bf16* W1_t = (__bf16*)alloc(WCM * 2);
  __bf16* W2_t = (__bf16*)alloc(WCM * 2);
  __bf16* h    = (__bf16*)alloc(ACT * 2);
  __bf16* qb   = (__bf16*)alloc(ACT * 2);
  __bf16* kb   = (__bf16*)alloc(ACT * 2);
  __bf16* vtb  = (__bf16*)alloc(ACT * 2);            // [C, S_PAD]
  __bf16* ctx  = (__bf16*)alloc(ACT * 2);
  float*  x1   = (float*)alloc((size_t)SEQ * NSTATE * 4);
  __bf16* h2   = (__bf16*)alloc(ACT * 2);
  __bf16* gmid = (__bf16*)alloc((size_t)SEQ_PAD * NMLP * 2);

  // 1) weights -> bf16, transposed [N,K]
  int gcc = (int)((WCC + 255) / 256), gcm = (int)((WCM + 255) / 256);
  convert_w_t<<<gcc, 256, 0, stream>>>(Wq, Wq_t, NSTATE, NSTATE);
  convert_w_t<<<gcc, 256, 0, stream>>>(Wk, Wk_t, NSTATE, NSTATE);
  convert_w_t<<<gcc, 256, 0, stream>>>(Wv, Wv_t, NSTATE, NSTATE);
  convert_w_t<<<gcc, 256, 0, stream>>>(Wo, Wo_t, NSTATE, NSTATE);
  convert_w_t<<<gcm, 256, 0, stream>>>(W1, W1_t, NSTATE, NMLP);
  convert_w_t<<<gcm, 256, 0, stream>>>(W2, W2_t, NMLP, NSTATE);

  // 2) LN1 -> h (bf16, padded)
  ln_kernel<<<SEQ_PAD, 256, 0, stream>>>(x, ln1g, ln1b, h);

  // 3) Q, K, V projections (V stored transposed)
  dim3 blk(128), grdC(MTILES / 2, NSTATE / 256), grdM(MTILES / 2, NMLP / 256);
  gemm_bf16_wmma<NSTATE, NSTATE, MODE_BF16>
      <<<grdC, blk, 0, stream>>>(h, Wq_t, bq, qb, nullptr);
  gemm_bf16_wmma<NSTATE, NSTATE, MODE_BF16>
      <<<grdC, blk, 0, stream>>>(h, Wk_t, nullptr, kb, nullptr);
  gemm_bf16_wmma<NSTATE, NSTATE, MODE_BF16_T>
      <<<grdC, blk, 0, stream>>>(h, Wv_t, bv, vtb, nullptr);

  // 4) flash attention: 20 heads * 94 q-tiles = 1880 waves = 470 blocks
  attn_kernel<<<(NHEAD * MTILES) / 4, 128, 0, stream>>>(qb, kb, vtb, ctx);

  // 5) output projection + residual -> x1 (fp32)
  gemm_bf16_wmma<NSTATE, NSTATE, MODE_F32_RES>
      <<<grdC, blk, 0, stream>>>(ctx, Wo_t, bo, x1, x);

  // 6) LN2 -> h2
  ln_kernel<<<SEQ_PAD, 256, 0, stream>>>(x1, ln2g, ln2b, h2);

  // 7) MLP up + exact GELU
  gemm_bf16_wmma<NSTATE, NMLP, MODE_GELU_BF16>
      <<<grdM, blk, 0, stream>>>(h2, W1_t, b1, gmid, nullptr);

  // 8) MLP down + residual -> d_out (fp32)
  gemm_bf16_wmma<NMLP, NSTATE, MODE_F32_RES>
      <<<grdC, blk, 0, stream>>>(gmid, W2_t, b2, out, x1);
}
